// SSM_12369505812864
// MI455X (gfx1250) — matrix operand order repaired and checked by reading, not compiled, not run on hardware
//
#include <hip/hip_runtime.h>
#include <math.h>

// ---------------------------------------------------------------------------
// SSM pipeline for MI455X (gfx1250, wave32, WMMA).
//   CVT  :  fp32 -> bf16 copies of u / Bmat / Cmat / Dmat
//   GEMM1:  bu[l*16+b][n] = sum_d u[b][l][d] * Bmat[n][d]        (bf16 WMMA)
//   SCAN :  3-pass chunked linear recurrence (32 chunks x 64 steps)
//           -> massively parallel, xs emitted as bf16
//   FUSED:  y = xs.C^T + u.D^T  -> GELU(erf) -> LayerNorm -> out[b][l][o]
//           (16-wave workgroup owns a full 16x1024 row tile; y never hits HBM)
// ---------------------------------------------------------------------------

typedef __attribute__((ext_vector_type(16))) __bf16 v16bf;
typedef __attribute__((ext_vector_type(8)))  float  v8f;

#define BATCH 16
#define SEQ   2048
#define KDIM  1024   // D_IN == N_STATE == D_OUT == 1024
#define MROWS (SEQ * BATCH)  // 32768
#define CHUNK 64
#define NCHUNK (SEQ / CHUNK) // 32
#define BN (BATCH * KDIM)    // 16384

// fp32 -> bf16, round-to-nearest-even
__device__ __forceinline__ unsigned short f32_to_bf16(float f) {
  unsigned int u = __float_as_uint(f);
  unsigned int r = u + 0x7FFFu + ((u >> 16) & 1u);
  return (unsigned short)(r >> 16);
}

__device__ __forceinline__ float neg_softplus(float ad) {
  return -((ad > 20.0f) ? ad : log1pf(expf(ad)));
}

__global__ __launch_bounds__(256)
void cvt_bf16_kernel(const float* __restrict__ in, unsigned short* __restrict__ out, int n) {
  int i = blockIdx.x * blockDim.x + threadIdx.x;
  int stride = gridDim.x * blockDim.x;
  for (; i < n; i += stride) out[i] = f32_to_bf16(in[i]);
}

// ---------------------------------------------------------------------------
// WMMA fragment loaders. Both operands stored K-major with K stride = 1024.
// A (16x32, bf16):  lane(m = lane&15, h = lane>>4), elem j -> K = 16*(j/8)+8h+(j%8)
//   => two contiguous 8-element (16B) chunks at k0+8h and k0+16+8h.
// B (32x16, bf16):  lane(n = lane&15, h = lane>>4), elem j -> K = 16*(j/8)+8h+(j%8)
//   => two contiguous 8-element (16B) chunks at k0+8h and k0+16+8h.
// ---------------------------------------------------------------------------
union Frag { v16bf v; uint4 q[2]; };

__device__ __forceinline__ Frag load_a_frag(const unsigned short* __restrict__ A,
                                            int rowMap, int m0, int k0, int lane) {
  int m = lane & 15, h = lane >> 4;
  int mg = m0 + m;
  // rowMap==1: output row m = l*16+b maps to u stored [b][l][:] ([B,L,D] layout)
  size_t row = rowMap ? ((size_t)(mg & (BATCH - 1)) * SEQ + (size_t)(mg >> 4))
                      : (size_t)mg;
  const unsigned short* p = A + row * (size_t)KDIM + (size_t)(k0 + 8 * h);
  Frag f;
  f.q[0] = *reinterpret_cast<const uint4*>(p);
  f.q[1] = *reinterpret_cast<const uint4*>(p + 16);
  __builtin_prefetch(p + 64, 0, 1);  // two K-blocks ahead -> global_prefetch_b8
  return f;
}

__device__ __forceinline__ Frag load_b_frag(const unsigned short* __restrict__ W,
                                            int n0, int k0, int lane) {
  int n = lane & 15, h = lane >> 4;
  const unsigned short* p = W + (size_t)(n0 + n) * (size_t)KDIM + (size_t)(k0 + 8 * h);
  Frag f;
  f.q[0] = *reinterpret_cast<const uint4*>(p);
  f.q[1] = *reinterpret_cast<const uint4*>(p + 16);
  return f;
}

__device__ __forceinline__ v8f wmma_bf16(const Frag& a, const Frag& b, v8f c) {
  return __builtin_amdgcn_wmma_f32_16x16x32_bf16(false, a.v, false, b.v,
                                                 (short)0, c, false, false);
}

// ---------------------------------------------------------------------------
// GEMM1: 4 waves/WG, 16(M) x 256(N) tile; wave tile 16x64 (4 accums).
// K loop unrolled x2; all fragment loads issued ahead of WMMAs so the
// scheduler overlaps VMEM with the matrix pipe (partial loadcnt waits).
// ---------------------------------------------------------------------------
__global__ __launch_bounds__(128)
void wmma_gemm1_kernel(const unsigned short* __restrict__ A,
                       const unsigned short* __restrict__ W,
                       float* __restrict__ Out) {
  const int lane = threadIdx.x & 31;
  const int wave = threadIdx.x >> 5;
  const int m0 = blockIdx.x * 16;
  const int n0 = blockIdx.y * 256 + wave * 64;

  const v8f zero = {0.f, 0.f, 0.f, 0.f, 0.f, 0.f, 0.f, 0.f};
  v8f acc[4] = {zero, zero, zero, zero};

  for (int k0 = 0; k0 < KDIM; k0 += 64) {
    Frag a0 = load_a_frag(A, 1, m0, k0, lane);
    Frag b0[4];
#pragma unroll
    for (int t = 0; t < 4; ++t) b0[t] = load_b_frag(W, n0 + 16 * t, k0, lane);
    Frag a1 = load_a_frag(A, 1, m0, k0 + 32, lane);
    Frag b1[4];
#pragma unroll
    for (int t = 0; t < 4; ++t) b1[t] = load_b_frag(W, n0 + 16 * t, k0 + 32, lane);
#pragma unroll
    for (int t = 0; t < 4; ++t) acc[t] = wmma_bf16(a0, b0[t], acc[t]);
#pragma unroll
    for (int t = 0; t < 4; ++t) acc[t] = wmma_bf16(a1, b1[t], acc[t]);
  }

  // C/D layout: VGPR r, lanes 0-15 -> M=r, lanes 16-31 -> M=8+r; N = lane&15.
  const int col = lane & 15;
  const int rbase = (lane >> 4) * 8;
#pragma unroll
  for (int t = 0; t < 4; ++t) {
#pragma unroll
    for (int r = 0; r < 8; ++r) {
      Out[(size_t)(m0 + rbase + r) * KDIM + (size_t)(n0 + 16 * t + col)] = acc[t][r];
    }
  }
}

// ---------------------------------------------------------------------------
// Chunked linear scan. x_t = a*x_{t-1} + bu_t is linear, so each 64-step chunk
// can run independently from x=0; carries recombine with a^64 afterwards.
// ---------------------------------------------------------------------------

// Pass 1: local scan per (chunk, b, n) from x=0; emit chunk-final value only.
__global__ __launch_bounds__(256)
void scan_pass1_kernel(const float* __restrict__ bu, const float* __restrict__ A_diag,
                       float* __restrict__ chunkFinal) {
  int t = blockIdx.x * blockDim.x + threadIdx.x;  // 0 .. NCHUNK*BN-1
  int n = t & (KDIM - 1);
  int b = (t >> 10) & (BATCH - 1);
  int c = t >> 14;
  const float a = neg_softplus(A_diag[n]);
  const size_t off = (size_t)b * KDIM + (size_t)n;
  const int l0 = c * CHUNK;
  float x = 0.f;
  for (int i = 0; i < CHUNK; i += 4) {
    float v0 = bu[(size_t)(l0 + i + 0) * BN + off];
    float v1 = bu[(size_t)(l0 + i + 1) * BN + off];
    float v2 = bu[(size_t)(l0 + i + 2) * BN + off];
    float v3 = bu[(size_t)(l0 + i + 3) * BN + off];
    x = fmaf(a, x, v0);
    x = fmaf(a, x, v1);
    x = fmaf(a, x, v2);
    x = fmaf(a, x, v3);
  }
  chunkFinal[t] = x;  // layout [c][b][n]
}

// Pass 2: per (b, n), serial scan of 32 chunk carries: carry' = a^64*carry + final.
__global__ __launch_bounds__(256)
void scan_pass2_kernel(const float* __restrict__ chunkFinal, const float* __restrict__ A_diag,
                       const float* __restrict__ h0, float* __restrict__ carryIn) {
  int t = blockIdx.x * blockDim.x + threadIdx.x;  // 0 .. BN-1
  int n = t & (KDIM - 1);
  const float a = neg_softplus(A_diag[n]);
  float a64 = a;
#pragma unroll
  for (int i = 0; i < 6; ++i) a64 *= a64;  // a^64
  float carry = h0[n];
  for (int c = 0; c < NCHUNK; ++c) {
    carryIn[(size_t)c * BN + (size_t)t] = carry;
    carry = fmaf(a64, carry, chunkFinal[(size_t)c * BN + (size_t)t]);
  }
}

// Pass 3: re-run each chunk from its true incoming carry; stream xs out as bf16.
__global__ __launch_bounds__(256)
void scan_pass3_kernel(const float* __restrict__ bu, const float* __restrict__ A_diag,
                       const float* __restrict__ carryIn, unsigned short* __restrict__ xs_bf) {
  int t = blockIdx.x * blockDim.x + threadIdx.x;  // 0 .. NCHUNK*BN-1
  int n = t & (KDIM - 1);
  int b = (t >> 10) & (BATCH - 1);
  int c = t >> 14;
  const float a = neg_softplus(A_diag[n]);
  const size_t off = (size_t)b * KDIM + (size_t)n;
  const int l0 = c * CHUNK;
  float x = carryIn[t];
  for (int i = 0; i < CHUNK; i += 4) {
    size_t i0 = (size_t)(l0 + i + 0) * BN + off;
    size_t i1 = (size_t)(l0 + i + 1) * BN + off;
    size_t i2 = (size_t)(l0 + i + 2) * BN + off;
    size_t i3 = (size_t)(l0 + i + 3) * BN + off;
    float v0 = bu[i0], v1 = bu[i1], v2 = bu[i2], v3 = bu[i3];
    x = fmaf(a, x, v0); xs_bf[i0] = f32_to_bf16(x);
    x = fmaf(a, x, v1); xs_bf[i1] = f32_to_bf16(x);
    x = fmaf(a, x, v2); xs_bf[i2] = f32_to_bf16(x);
    x = fmaf(a, x, v3); xs_bf[i3] = f32_to_bf16(x);
  }
}

// ---------------------------------------------------------------------------
// Fused GEMM2 + GELU + LayerNorm.
// 512 threads = 16 waves; wave w owns cols [w*64, w*64+64) of a 16x1024 tile.
// Dual-K accumulation: y = xs.C^T + u.D^T. Then exact-erf GELU, cross-wave LDS
// reduction for per-row mean/var, normalize, nan_to_num, store to out[b][l][o].
// ---------------------------------------------------------------------------
__global__ __launch_bounds__(512, 1)
void wmma_gemm2_ln_kernel(const unsigned short* __restrict__ Axs,
                          const unsigned short* __restrict__ Wc,
                          const unsigned short* __restrict__ Au,
                          const unsigned short* __restrict__ Wd,
                          const float* __restrict__ gamma,
                          const float* __restrict__ beta,
                          float* __restrict__ out) {
  __shared__ float s_sum[16][256];
  __shared__ float s_sq[16][256];
  __shared__ float s2s[16][32];
  __shared__ float s2q[16][32];
  __shared__ float s_mean[16];
  __shared__ float s_inv[16];

  const int tid = threadIdx.x;
  const int lane = tid & 31;
  const int wave = tid >> 5;           // 0..15
  const int m0 = blockIdx.x * 16;
  const int n0 = wave * 64;

  const v8f zero = {0.f, 0.f, 0.f, 0.f, 0.f, 0.f, 0.f, 0.f};
  v8f acc[4] = {zero, zero, zero, zero};

  for (int k0 = 0; k0 < KDIM; k0 += 32) {
    {  // phase 1: xs . C^T
      Frag a = load_a_frag(Axs, 0, m0, k0, lane);
      Frag b[4];
#pragma unroll
      for (int t = 0; t < 4; ++t) b[t] = load_b_frag(Wc, n0 + 16 * t, k0, lane);
#pragma unroll
      for (int t = 0; t < 4; ++t) acc[t] = wmma_bf16(a, b[t], acc[t]);
    }
    {  // phase 2: u . D^T (accumulate)
      Frag a = load_a_frag(Au, 1, m0, k0, lane);
      Frag b[4];
#pragma unroll
      for (int t = 0; t < 4; ++t) b[t] = load_b_frag(Wd, n0 + 16 * t, k0, lane);
#pragma unroll
      for (int t = 0; t < 4; ++t) acc[t] = wmma_bf16(a, b[t], acc[t]);
    }
  }

  // Exact GELU in-place + per-lane row partial sums (8 rows per lane).
  const int rbase = (lane >> 4) * 8;          // rows rbase..rbase+7
  const int cidx = wave * 16 + (lane & 15);   // 0..255, unique per row contributor
  float rs[8], rq[8];
#pragma unroll
  for (int r = 0; r < 8; ++r) { rs[r] = 0.f; rq[r] = 0.f; }
#pragma unroll
  for (int t = 0; t < 4; ++t) {
#pragma unroll
    for (int r = 0; r < 8; ++r) {
      float v = acc[t][r];
      float g = 0.5f * v * (1.0f + erff(v * 0.70710678118654752f));
      acc[t][r] = g;
      rs[r] += g;
      rq[r] += g * g;
    }
  }
#pragma unroll
  for (int r = 0; r < 8; ++r) {
    s_sum[rbase + r][cidx] = rs[r];
    s_sq[rbase + r][cidx] = rq[r];
  }
  __syncthreads();

  // Stage 1: 32 threads per row fold 256 partials -> 32.
  {
    int row = tid >> 5, j = tid & 31;
    float p = 0.f, q = 0.f;
#pragma unroll
    for (int i = 0; i < 8; ++i) {
      p += s_sum[row][j + 32 * i];
      q += s_sq[row][j + 32 * i];
    }
    s2s[row][j] = p;
    s2q[row][j] = q;
  }
  __syncthreads();

  // Stage 2: one thread per row -> mean / inv-std.
  if (tid < 16) {
    float p = 0.f, q = 0.f;
#pragma unroll
    for (int j = 0; j < 32; ++j) { p += s2s[tid][j]; q += s2q[tid][j]; }
    float mean = p * (1.0f / KDIM);
    float var = q * (1.0f / KDIM) - mean * mean;  // population var (ddof=0)
    s_mean[tid] = mean;
    s_inv[tid] = rsqrtf(var + 1e-5f);
  }
  __syncthreads();

  // Normalize + affine + nan_to_num, store to out[b][l][o].
  const int colb = lane & 15;
#pragma unroll
  for (int t = 0; t < 4; ++t) {
    const int col = n0 + 16 * t + colb;
    const float gm = gamma[col];
    const float bt = beta[col];
#pragma unroll
    for (int r = 0; r < 8; ++r) {
      const int row = rbase + r;
      const int m = m0 + row;
      const int b = m & (BATCH - 1);
      const int l = m >> 4;
      float rr = (acc[t][r] - s_mean[row]) * s_inv[row] * gm + bt;
      if (rr != rr) rr = 0.0f;                    // nan -> 0
      rr = fminf(fmaxf(rr, -1.0e6f), 1.0e6f);     // +/-inf -> +/-CLAMP
      out[((size_t)b * SEQ + (size_t)l) * KDIM + (size_t)col] = rr;
    }
  }
}

extern "C" void kernel_launch(void* const* d_in, const int* in_sizes, int n_in,
                              void* d_out, int out_size, void* d_ws, size_t ws_size,
                              hipStream_t stream) {
  (void)in_sizes; (void)n_in; (void)out_size; (void)ws_size;
  const float* u      = (const float*)d_in[0];
  const float* A_diag = (const float*)d_in[1];
  const float* Bmat   = (const float*)d_in[2];
  const float* Cmat   = (const float*)d_in[3];
  const float* Dmat   = (const float*)d_in[4];
  const float* h0     = (const float*)d_in[5];
  const float* gamma  = (const float*)d_in[6];
  const float* beta   = (const float*)d_in[7];
  float* out = (float*)d_out;

  // Workspace carve-out (256B aligned slabs).
  char* w = (char*)d_ws;
  auto carve = [&](size_t bytes) {
    char* p = w;
    w += (bytes + 255) & ~(size_t)255;
    return p;
  };
  unsigned short* uBf   = (unsigned short*)carve((size_t)BATCH * SEQ * KDIM * 2);  // 64 MB
  unsigned short* Bbf   = (unsigned short*)carve((size_t)KDIM * KDIM * 2);         //  2 MB
  unsigned short* Cbf   = (unsigned short*)carve((size_t)KDIM * KDIM * 2);
  unsigned short* Dbf   = (unsigned short*)carve((size_t)KDIM * KDIM * 2);
  unsigned short* xsBf  = (unsigned short*)carve((size_t)MROWS * KDIM * 2);        // 64 MB
  float*          bu    = (float*)carve((size_t)MROWS * KDIM * 4);                 // 128 MB
  float*          cfin  = (float*)carve((size_t)NCHUNK * BN * 4);                  //  2 MB
  float*          cin   = (float*)carve((size_t)NCHUNK * BN * 4);                  //  2 MB

  // 1) fp32 -> bf16 conversions
  cvt_bf16_kernel<<<2048, 256, 0, stream>>>(u, uBf, BATCH * SEQ * KDIM);
  cvt_bf16_kernel<<<512, 256, 0, stream>>>(Bmat, Bbf, KDIM * KDIM);
  cvt_bf16_kernel<<<512, 256, 0, stream>>>(Cmat, Cbf, KDIM * KDIM);
  cvt_bf16_kernel<<<512, 256, 0, stream>>>(Dmat, Dbf, KDIM * KDIM);

  // 2) GEMM1: bu[l*16+b][n] = u_row(b,l) . Bmat[n]
  wmma_gemm1_kernel<<<dim3(MROWS / 16, KDIM / 256), 128, 0, stream>>>(uBf, Bbf, bu);

  // 3) chunked linear scan: local chunks -> carry scan -> final chunks (xs bf16)
  scan_pass1_kernel<<<(NCHUNK * BN) / 256, 256, 0, stream>>>(bu, A_diag, cfin);
  scan_pass2_kernel<<<BN / 256, 256, 0, stream>>>(cfin, A_diag, h0, cin);
  scan_pass3_kernel<<<(NCHUNK * BN) / 256, 256, 0, stream>>>(bu, A_diag, cin, xsBf);

  // 4) fused GEMM2 (dual-K) + GELU + LayerNorm -> out[b][l][o]
  wmma_gemm2_ln_kernel<<<MROWS / 16, 512, 0, stream>>>(xsBf, Cbf, uBf, Dbf,
                                                       gamma, beta, out);
}
